// MimiStreamingMultiheadAttention_3796751090426
// MI455X (gfx1250) — compile-verified, matrix operand order checked by reading, not run on hardware
//
#include <hip/hip_runtime.h>
#include <hip/hip_bf16.h>
#include <math.h>

// Problem constants (from reference)
#define EMB   1024
#define NH    16
#define DH    64
#define TLEN  2048
#define BSZ   2
#define WIN   250

typedef __attribute__((ext_vector_type(16))) __bf16          v16bf;
typedef __attribute__((ext_vector_type(8)))  float           v8f;
typedef __attribute__((ext_vector_type(16))) unsigned short  v16u;
typedef __attribute__((ext_vector_type(8)))  unsigned short  v8u;

__device__ __forceinline__ unsigned short f2bf(float f) {
    unsigned int u = __builtin_bit_cast(unsigned int, f);
    u += 0x7FFFu + ((u >> 16) & 1u);          // round-to-nearest-even
    return (unsigned short)(u >> 16);
}
__device__ __forceinline__ float bf2f(unsigned short h) {
    unsigned int u = ((unsigned int)h) << 16;
    return __builtin_bit_cast(float, u);
}
__device__ __forceinline__ v16bf mkfrag(v8u lo, v8u hi) {
    v16u v = __builtin_shufflevector(lo, hi, 0,1,2,3,4,5,6,7,8,9,10,11,12,13,14,15);
    return __builtin_bit_cast(v16bf, v);
}
__device__ __forceinline__ v8f vzero8() {
    v8f z;
#pragma unroll
    for (int i = 0; i < 8; ++i) z[i] = 0.0f;
    return z;
}
__device__ __forceinline__ v8f wmma_bf16(v16bf a, v16bf b, v8f c) {
    return __builtin_amdgcn_wmma_f32_16x16x32_bf16(false, a, false, b, (short)0, c, false, false);
}

// CDNA5 async global->LDS copy (ASYNCcnt path). LDS byte address = low 32 bits
// of the flat pointer (ISA 10.2).
__device__ __forceinline__ void async_copy_b128(void* lds_ptr, const void* gptr) {
    unsigned lds_off = (unsigned)(unsigned long long)lds_ptr;
    asm volatile("global_load_async_to_lds_b128 %0, %1, off"
                 :: "v"(lds_off), "v"(gptr)
                 : "memory");
}
// Async loads complete in order: waiting <=4 outstanding guarantees the 4
// older (current-tile) transfers have landed while the 4 newer (next-tile)
// transfers remain in flight.
__device__ __forceinline__ void wait_async_le(int n) {
    if (n == 0) asm volatile("s_wait_asynccnt 0x0" ::: "memory");
    else        asm volatile("s_wait_asynccnt 0x4" ::: "memory");
}

// ---------------------------------------------------------------------------
// f32 -> bf16 conversion (vectorized x4)
// ---------------------------------------------------------------------------
__global__ void cvt_f32_bf16_kernel(const float* __restrict__ src,
                                    unsigned short* __restrict__ dst, int n4) {
    int i = blockIdx.x * blockDim.x + threadIdx.x;
    if (i >= n4) return;
    float4 f = ((const float4*)src)[i];
    unsigned long long p = (unsigned long long)f2bf(f.x)
                         | ((unsigned long long)f2bf(f.y) << 16)
                         | ((unsigned long long)f2bf(f.z) << 32)
                         | ((unsigned long long)f2bf(f.w) << 48);
    ((unsigned long long*)dst)[i] = p;
}

// ---------------------------------------------------------------------------
// Tiled bf16 WMMA GEMM: C[M][N] = A[M][1024] * Bw[N][1024]^T
// Block tile 128(M) x 128(N), 8 waves (4 M x 2 N), wave tile 32x64:
// 8 WMMAs per k-step per wave. Double-buffered async global->LDS staging
// (ASYNCcnt pipelining: next tile in flight while current tile computes).
// mode 0: scatter to Q,K [B,H,T,D] and V transposed [B,H,D,T] (bf16)
// mode 1: write f32 to Fo[M][1024]
// ---------------------------------------------------------------------------
__global__ __launch_bounds__(256)
void gemm_bf16_wmma_kernel(const unsigned short* __restrict__ A,
                           const unsigned short* __restrict__ Bw,
                           int mode,
                           unsigned short* __restrict__ Qo,
                           unsigned short* __restrict__ Ko,
                           unsigned short* __restrict__ Vto,
                           float* __restrict__ Fo) {
    __shared__ __align__(16) unsigned short As[2][128][40];
    __shared__ __align__(16) unsigned short Bs[2][128][40];

    const int tid   = threadIdx.x;
    const int lane  = tid & 31;
    const int wid   = tid >> 5;
    const int waveM = wid & 3;            // 0..3 -> M offset *32
    const int waveN = wid >> 2;           // 0..1 -> N offset *64
    const int m0    = blockIdx.y * 128;
    const int n0    = blockIdx.x * 128;

    // staging decomposition: 128 rows x 4 chunks(8 elem) = 512 chunks, 2/thread
    const int r0 = tid >> 1;              // 0..127
    const int c0 = (tid & 1) * 8;         // chunks 0,1
    const int c1 = c0 + 16;               // chunks 2,3

    const int col  = lane & 15;
    const int base = (lane >> 4) * 8;     // 0 or 8 (A/B lane K-half pattern)
    const int hf   = lane >> 4;

    const unsigned short* gA0 = A  + (m0 + r0) * 1024 + c0;
    const unsigned short* gA1 = A  + (m0 + r0) * 1024 + c1;
    const unsigned short* gB0 = Bw + (n0 + r0) * 1024 + c0;
    const unsigned short* gB1 = Bw + (n0 + r0) * 1024 + c1;
    void* lA0[2] = { &As[0][r0][c0], &As[1][r0][c0] };
    void* lA1[2] = { &As[0][r0][c1], &As[1][r0][c1] };
    void* lB0[2] = { &Bs[0][r0][c0], &Bs[1][r0][c0] };
    void* lB1[2] = { &Bs[0][r0][c1], &Bs[1][r0][c1] };

    v8f acc[2][4];
#pragma unroll
    for (int i = 0; i < 2; ++i)
#pragma unroll
        for (int j = 0; j < 4; ++j) acc[i][j] = vzero8();

    // prologue: tile 0 in flight
    async_copy_b128(lA0[0], gA0);
    async_copy_b128(lA1[0], gA1);
    async_copy_b128(lB0[0], gB0);
    async_copy_b128(lB1[0], gB1);

    for (int it = 0; it < 32; ++it) {         // 32 k-steps of 32
        const int cur = it & 1;
        const int nxt = cur ^ 1;
        if (it < 31) {
            const int koff = (it + 1) * 32;
            async_copy_b128(lA0[nxt], gA0 + koff);
            async_copy_b128(lA1[nxt], gA1 + koff);
            async_copy_b128(lB0[nxt], gB0 + koff);
            async_copy_b128(lB1[nxt], gB1 + koff);
            wait_async_le(4);                 // current tile landed, next in flight
        } else {
            wait_async_le(0);
        }
        __syncthreads();                      // all waves' current-tile data visible

        v16bf a[2];
#pragma unroll
        for (int i = 0; i < 2; ++i) {
            const int mrow = waveM * 32 + i * 16 + col;
            a[i] = mkfrag(*(const v8u*)&As[cur][mrow][base],
                          *(const v8u*)&As[cur][mrow][base + 16]);
        }
#pragma unroll
        for (int j = 0; j < 4; ++j) {
            const int ncol = waveN * 64 + j * 16 + col;
            v16bf b = mkfrag(*(const v8u*)&Bs[cur][ncol][base],
                             *(const v8u*)&Bs[cur][ncol][base + 16]);
#pragma unroll
            for (int i = 0; i < 2; ++i)
                acc[i][j] = wmma_bf16(a[i], b, acc[i][j]);
        }
        __syncthreads();                      // reads done before buffer reuse
    }

#pragma unroll
    for (int i = 0; i < 2; ++i) {
#pragma unroll
        for (int j = 0; j < 4; ++j) {
#pragma unroll
            for (int r = 0; r < 8; ++r) {
                int gm = m0 + waveM * 32 + i * 16 + r + 8 * hf;
                int gn = n0 + waveN * 64 + j * 16 + col;
                float v = acc[i][j][r];
                if (mode == 0) {
                    int which = gn >> 10;               // 0=q,1=k,2=v
                    int rem = gn & 1023;
                    int h = rem >> 6, d = rem & 63;
                    int b = gm >> 11, t = gm & 2047;
                    unsigned short bv = f2bf(v);
                    if (which == 0)      Qo [(((b * 16 + h) * 2048 + t) << 6) + d] = bv;
                    else if (which == 1) Ko [(((b * 16 + h) * 2048 + t) << 6) + d] = bv;
                    else                 Vto[(((b * 16 + h) * 64 + d) << 11) + t] = bv;
                } else {
                    Fo[gm * 1024 + gn] = v;
                }
            }
        }
    }
}

// ---------------------------------------------------------------------------
// Interleaved-pair RoPE on bf16 Q,K in [B,H,T,D] layout
// ---------------------------------------------------------------------------
__global__ void rope_kernel(unsigned short* __restrict__ Q,
                            unsigned short* __restrict__ K) {
    int i = blockIdx.x * blockDim.x + threadIdx.x;   // over 32 * 2048 * 32
    if (i >= 32 * 2048 * 32) return;
    int d2 = i & 31;
    int t  = (i >> 5) & 2047;
    int bh = i >> 16;
    const float coef = -0.28782313663f;              // -ln(10000) * 2 / 64
    float freq = __expf((float)d2 * coef);
    float s, c;
    __sincosf((float)t * freq, &s, &c);
    int idx = ((bh * 2048 + t) << 6) + 2 * d2;
    {
        float xa = bf2f(Q[idx]), xb = bf2f(Q[idx + 1]);
        Q[idx]     = f2bf(xa * c - xb * s);
        Q[idx + 1] = f2bf(xa * s + xb * c);
    }
    {
        float xa = bf2f(K[idx]), xb = bf2f(K[idx + 1]);
        K[idx]     = f2bf(xa * c - xb * s);
        K[idx + 1] = f2bf(xa * s + xb * c);
    }
}

// ---------------------------------------------------------------------------
// Sliding-window flash attention, one wave per (b,h, 16-query tile).
// Per 32-key step: 4 WMMAs for S=QK^T (K=64 split), online softmax with
// 16-lane shfl reductions, P staged via LDS (C-layout -> A-layout), then
// 4 WMMAs for O += P * V (K=32), V pre-transposed [B,H,D,T].
// ---------------------------------------------------------------------------
__global__ __launch_bounds__(32)
void attn_kernel(const unsigned short* __restrict__ Q,
                 const unsigned short* __restrict__ K,
                 const unsigned short* __restrict__ Vt,
                 unsigned short* __restrict__ Xatt) {
    __shared__ __align__(16) unsigned short sP[16][40];

    const int lane = threadIdx.x & 31;
    const int q0   = blockIdx.x * 16;
    const int bh   = blockIdx.y;       // b*16 + h
    const int b    = bh >> 4;
    const int h    = bh & 15;

    const unsigned short* Qp = Q  + ((long)bh << 17);   // bh * T * D
    const unsigned short* Kp = K  + ((long)bh << 17);
    const unsigned short* Vp = Vt + ((long)bh << 17);   // [D][T]

    const int col  = lane & 15;
    const int hf   = lane >> 4;
    const int base = hf * 8;

    const unsigned short* qrow = Qp + (long)(q0 + col) * 64;
    v16bf aq0 = mkfrag(*(const v8u*)(qrow + base),      *(const v8u*)(qrow + base + 16));
    v16bf aq1 = mkfrag(*(const v8u*)(qrow + 32 + base), *(const v8u*)(qrow + 48 + base));

    v8f O[4];
#pragma unroll
    for (int nt = 0; nt < 4; ++nt) O[nt] = vzero8();
    float M[8], L[8];
#pragma unroll
    for (int r = 0; r < 8; ++r) { M[r] = -1.0e30f; L[r] = 0.0f; }

    int lo = q0 - (WIN - 1); if (lo < 0) lo = 0;
    const int s_begin = (lo >> 5) << 5;

    for (int s0 = s_begin; s0 <= q0 + 15; s0 += 32) {
        v8f S0 = vzero8(), S1 = vzero8();
        const unsigned short* krow0 = Kp + (long)(s0 + col) * 64;
        const unsigned short* krow1 = Kp + (long)(s0 + 16 + col) * 64;
        v16bf bk;
        bk = mkfrag(*(const v8u*)(krow0 + base),      *(const v8u*)(krow0 + base + 16));
        S0 = wmma_bf16(aq0, bk, S0);
        bk = mkfrag(*(const v8u*)(krow0 + 32 + base), *(const v8u*)(krow0 + 48 + base));
        S0 = wmma_bf16(aq1, bk, S0);
        bk = mkfrag(*(const v8u*)(krow1 + base),      *(const v8u*)(krow1 + base + 16));
        S1 = wmma_bf16(aq0, bk, S1);
        bk = mkfrag(*(const v8u*)(krow1 + 32 + base), *(const v8u*)(krow1 + 48 + base));
        S1 = wmma_bf16(aq1, bk, S1);

        float corr[8];
#pragma unroll
        for (int r = 0; r < 8; ++r) {
            const int tq = q0 + r + 8 * hf;
            const int d0 = tq - (s0 + col);
            const int d1 = tq - (s0 + 16 + col);
            float s0v = S0[r] * 0.125f;   // D^-0.5 = 1/8
            float s1v = S1[r] * 0.125f;
            if (d0 < 0 || d0 >= WIN) s0v = -3.0e38f;
            if (d1 < 0 || d1 >= WIN) s1v = -3.0e38f;
            float mloc = fmaxf(s0v, s1v);
#pragma unroll
            for (int msk = 1; msk < 16; msk <<= 1)
                mloc = fmaxf(mloc, __shfl_xor(mloc, msk, 16));
            float Mnew = fmaxf(M[r], mloc);
            float cr = __expf(M[r] - Mnew);
            float p0 = __expf(s0v - Mnew);
            float p1 = __expf(s1v - Mnew);
            float ls = p0 + p1;
#pragma unroll
            for (int msk = 1; msk < 16; msk <<= 1)
                ls += __shfl_xor(ls, msk, 16);
            L[r] = L[r] * cr + ls;
            M[r] = Mnew;
            corr[r] = cr;
            sP[r + 8 * hf][col]      = f2bf(p0);
            sP[r + 8 * hf][16 + col] = f2bf(p1);
        }
        __syncthreads();
        v16bf ap = mkfrag(*(const v8u*)&sP[col][base], *(const v8u*)&sP[col][base + 16]);
#pragma unroll
        for (int nt = 0; nt < 4; ++nt) {
#pragma unroll
            for (int r = 0; r < 8; ++r) O[nt][r] *= corr[r];
            const unsigned short* vrow = Vp + (long)(nt * 16 + col) * 2048 + s0;
            v16bf bv = mkfrag(*(const v8u*)(vrow + base), *(const v8u*)(vrow + 16 + base));
            O[nt] = wmma_bf16(ap, bv, O[nt]);
        }
        __syncthreads();
    }

#pragma unroll
    for (int nt = 0; nt < 4; ++nt) {
#pragma unroll
        for (int r = 0; r < 8; ++r) {
            int t = q0 + r + 8 * hf;
            float v = O[nt][r] / L[r];
            Xatt[(long)(b * 2048 + t) * 1024 + h * 64 + nt * 16 + col] = f2bf(v);
        }
    }
}

// ---------------------------------------------------------------------------
extern "C" void kernel_launch(void* const* d_in, const int* in_sizes, int n_in,
                              void* d_out, int out_size, void* d_ws, size_t ws_size,
                              hipStream_t stream) {
    const float* query = (const float*)d_in[0];   // [2,2048,1024]
    const float* wqkv  = (const float*)d_in[1];   // [3072,1024]
    const float* wout  = (const float*)d_in[2];   // [1024,1024]
    float* out = (float*)d_out;                   // [2,2048,1024]

    unsigned short* Xb   = (unsigned short*)d_ws;          // 4096*1024
    unsigned short* Wq   = Xb  + 4096 * 1024;              // 3072*1024
    unsigned short* Wo   = Wq  + 3072 * 1024;              // 1024*1024
    unsigned short* Qb   = Wo  + 1024 * 1024;              // [B,H,T,D]
    unsigned short* Kb   = Qb  + 4096 * 1024;              // [B,H,T,D]
    unsigned short* Vtb  = Kb  + 4096 * 1024;              // [B,H,D,T]
    unsigned short* Xatt = Vtb + 4096 * 1024;              // [B,T,E]

    const int nq4 = 4096 * 1024 / 4, nw4 = 3072 * 1024 / 4, no4 = 1024 * 1024 / 4;
    cvt_f32_bf16_kernel<<<(nq4 + 255) / 256, 256, 0, stream>>>(query, Xb, nq4);
    cvt_f32_bf16_kernel<<<(nw4 + 255) / 256, 256, 0, stream>>>(wqkv, Wq, nw4);
    cvt_f32_bf16_kernel<<<(no4 + 255) / 256, 256, 0, stream>>>(wout, Wo, no4);

    // QKV projection: M=4096, N=3072
    gemm_bf16_wmma_kernel<<<dim3(3072 / 128, 4096 / 128), 256, 0, stream>>>(
        Xb, Wq, 0, Qb, Kb, Vtb, nullptr);

    rope_kernel<<<(32 * 2048 * 32 + 255) / 256, 256, 0, stream>>>(Qb, Kb);

    attn_kernel<<<dim3(2048 / 16, BSZ * NH), 32, 0, stream>>>(Qb, Kb, Vtb, Xatt);

    // Output projection: M=4096, N=1024 -> f32 d_out
    gemm_bf16_wmma_kernel<<<dim3(1024 / 128, 4096 / 128), 256, 0, stream>>>(
        Xatt, Wo, 1, nullptr, nullptr, nullptr, out);
}
